// myNet_18889266168135
// MI455X (gfx1250) — matrix operand order repaired and checked by reading, not compiled, not run on hardware
//
#include <hip/hip_runtime.h>
#include <math.h>

typedef _Float16 v16h __attribute__((ext_vector_type(16)));
typedef float    v8f  __attribute__((ext_vector_type(8)));

#define B_TOTAL 16384
#define N_CAT   26
#define N_CTS   13
#define NF      39          // N_CAT + N_CTS
#define VOC     100000
#define DIM     64
#define HID     512
#define KTOT    (NF * DIM)  // 2496
#define ROWS    32          // batch rows per block
#define LDS_STR (KTOT + 16) // padded stride in halves (2512)

union V16HQ { v16h v; uint4 q[2]; };
union H2U   { _Float16 h[2]; unsigned u; };
union H4U2  { _Float16 h[4]; uint2 u; };

// ---- pre-pass: W1 (512x2496 f32) -> f16 in workspace, read once per block after
__launch_bounds__(256)
__global__ void w1_to_f16(const float* __restrict__ W1, _Float16* __restrict__ W1h) {
    const int i = (blockIdx.x * 256 + threadIdx.x) * 4; // HID*KTOT = 1277952 = 1248*256*4
    const float4 v = *(const float4*)(W1 + i);
    H4U2 p;
    p.h[0] = (_Float16)v.x; p.h[1] = (_Float16)v.y;
    p.h[2] = (_Float16)v.z; p.h[3] = (_Float16)v.w;
    *(uint2*)(W1h + i) = p.u;
}

template <bool W16>
__launch_bounds__(256)
__global__ void dlrm_fused(const float* __restrict__ x,
                           const float* __restrict__ emb,
                           const float* __restrict__ ctsW,
                           const float* __restrict__ ctsB,
                           const float* __restrict__ W1,
                           const _Float16* __restrict__ W1h,
                           const float* __restrict__ b1,
                           const float* __restrict__ W2,
                           const float* __restrict__ b2,
                           float* __restrict__ out)
{
    __shared__ _Float16 sh[ROWS * LDS_STR];
    __shared__ float    sout[ROWS];

    const int tid  = threadIdx.x;
    const int row0 = blockIdx.x * ROWS;

    if (tid < ROWS) sout[tid] = 0.0f;

    // ---------------- Phase 1: build (ROWS x KTOT) f16 activation tile in LDS.
    // Two consecutive elements per iteration (same row & feature): b64 gather,
    // packed cvt, b32 LDS store. 39936 = 156*256 exactly -> no divergence.
    const int total2 = ROWS * KTOT / 2;
    for (int e = tid; e < total2; e += 256) {
        const int e2 = e * 2;
        const int r = e2 / KTOT;
        const int c = e2 - r * KTOT;
        const int f = c >> 6;
        const int d = c & 63;          // even
        const int grow = row0 + r;
        float v0, v1;
        if (f < N_CAT) {
            const int idx = (int)x[grow * NF + f];
            const float2 ev =
                *(const float2*)(emb + ((size_t)f * VOC + (size_t)idx) * DIM + d);
            v0 = ev.x; v1 = ev.y;
            if (f > 0) { v0 = fmaxf(v0, 0.0f); v1 = fmaxf(v1, 0.0f); }
        } else {
            const int fc = f - N_CAT;
            const float xv = x[grow * NF + f];
            const float2 wv = *(const float2*)(ctsW + fc * DIM + d);
            const float2 bv = *(const float2*)(ctsB + fc * DIM + d);
            v0 = fmaxf(xv * wv.x + bv.x, 0.0f);
            v1 = fmaxf(xv * wv.y + bv.y, 0.0f);
        }
        H2U p; p.h[0] = (_Float16)v0; p.h[1] = (_Float16)v1;
        *(unsigned*)&sh[r * LDS_STR + c] = p.u;
    }
    __syncthreads();

    // ---------------- Phase 2: GEMM1 via WMMA f16 -> f32.
    // 8 waves; wave w owns columns [w*64, w*64+64) (4 N-tiles), both M-tiles,
    // so W1 is read exactly once per block (L2-resident).
    const int wid  = tid >> 5;
    const int lane = tid & 31;
    const int hi   = lane >> 4;
    const int lr   = lane & 15;

    v8f acc[2][4] = {};
    const int ncol = wid * 64 + lr;

    for (int kb = 0; kb < KTOT; kb += 32) {
        // A frags: lane = row M=lr;
        // halves 0..15 at K=kb+hi*16 .. +15 contiguous (matches B).
        V16HQ a[2];
#pragma unroll
        for (int mt = 0; mt < 2; ++mt) {
            const uint4* ap =
                (const uint4*)&sh[(mt * 16 + lr) * LDS_STR + kb + hi * 16];
            a[mt].q[0] = ap[0];
            a[mt].q[1] = ap[1];
        }
#pragma unroll
        for (int nt = 0; nt < 4; ++nt) {
            // B frag: lane = column N, K = kb+hi*16 .. +15 contiguous.
            const int n = ncol + nt * 16;
            v16h b;
            if constexpr (W16) {
                const uint4* wp =
                    (const uint4*)(W1h + (size_t)n * KTOT + kb + hi * 16);
                V16HQ bb; bb.q[0] = wp[0]; bb.q[1] = wp[1];
                b = bb.v;
            } else {
                const float4* wp =
                    (const float4*)(W1 + (size_t)n * KTOT + kb + hi * 16);
                const float4 q0 = wp[0], q1 = wp[1], q2 = wp[2], q3 = wp[3];
                b[0]=(_Float16)q0.x;  b[1]=(_Float16)q0.y;  b[2]=(_Float16)q0.z;  b[3]=(_Float16)q0.w;
                b[4]=(_Float16)q1.x;  b[5]=(_Float16)q1.y;  b[6]=(_Float16)q1.z;  b[7]=(_Float16)q1.w;
                b[8]=(_Float16)q2.x;  b[9]=(_Float16)q2.y;  b[10]=(_Float16)q2.z; b[11]=(_Float16)q2.w;
                b[12]=(_Float16)q3.x; b[13]=(_Float16)q3.y; b[14]=(_Float16)q3.z; b[15]=(_Float16)q3.w;
            }
            acc[0][nt] = __builtin_amdgcn_wmma_f32_16x16x32_f16(
                false, a[0].v, false, b, (short)0, acc[0][nt], false, false);
            acc[1][nt] = __builtin_amdgcn_wmma_f32_16x16x32_f16(
                false, a[1].v, false, b, (short)0, acc[1][nt], false, false);
        }
    }

    // ---------------- Phase 3: bias + relu + dot with W2, reduce per row.
    float b1v[4], w2v[4];
#pragma unroll
    for (int nt = 0; nt < 4; ++nt) {
        const int n = ncol + nt * 16;
        b1v[nt] = b1[n];
        w2v[nt] = W2[n];
    }
#pragma unroll
    for (int mt = 0; mt < 2; ++mt) {
#pragma unroll
        for (int i = 0; i < 8; ++i) {
            float p = 0.0f;
#pragma unroll
            for (int nt = 0; nt < 4; ++nt) {
                const float hv = fmaxf(acc[mt][nt][i] + b1v[nt], 0.0f);
                p += hv * w2v[nt];
            }
            // C layout: VGPR i, lanes 0-15 -> M=i, lanes 16-31 -> M=i+8
            atomicAdd(&sout[mt * 16 + i + hi * 8], p);
        }
    }
    __syncthreads();

    if (tid < ROWS)
        out[row0 + tid] = expf(sout[tid] + b2[0]);
}

extern "C" void kernel_launch(void* const* d_in, const int* in_sizes, int n_in,
                              void* d_out, int out_size, void* d_ws, size_t ws_size,
                              hipStream_t stream) {
    const float* x    = (const float*)d_in[0];
    const float* emb  = (const float*)d_in[1];
    const float* ctsW = (const float*)d_in[2];
    const float* ctsB = (const float*)d_in[3];
    const float* W1   = (const float*)d_in[4];
    const float* b1   = (const float*)d_in[5];
    const float* W2   = (const float*)d_in[6];
    const float* b2   = (const float*)d_in[7];

    const size_t need = (size_t)HID * KTOT * sizeof(_Float16); // 2.55 MB
    if (ws_size >= need) {
        _Float16* W1h = (_Float16*)d_ws;
        w1_to_f16<<<(HID * KTOT / 4) / 256, 256, 0, stream>>>(W1, W1h);
        dlrm_fused<true><<<B_TOTAL / ROWS, 256, 0, stream>>>(
            x, emb, ctsW, ctsB, W1, W1h, b1, W2, b2, (float*)d_out);
    } else {
        dlrm_fused<false><<<B_TOTAL / ROWS, 256, 0, stream>>>(
            x, emb, ctsW, ctsB, W1, nullptr, b1, W2, b2, (float*)d_out);
    }
}
